// lct_20177756356742
// MI455X (gfx1250) — matrix operand order repaired and checked by reading, not compiled, not run on hardware
//
#include <hip/hip_runtime.h>

typedef __attribute__((ext_vector_type(16))) _Float16 v16h;
typedef __attribute__((ext_vector_type(8)))  float    v8f;
typedef __attribute__((ext_vector_type(2)))  float    v2f;

#define NTOT     33554432    // 512*256*256
#define PADBYTES 268435456ULL

#if defined(__has_builtin)
# if __has_builtin(__builtin_amdgcn_wmma_f32_16x16x4_f32)
#  define LCT_HAVE_WMMA_F32 1
# endif
#endif
#ifndef LCT_HAVE_WMMA_F32
# define LCT_HAVE_WMMA_F32 0
#endif

// ---------------------------------------------------------------------------
// K0: fold diffuse-falloff z^4 into the temporal resampling matrix (fp32)
// ---------------------------------------------------------------------------
__global__ __launch_bounds__(256)
void lct_prep_mtx(const float* __restrict__ mtx, const float* __restrict__ gz,
                  float* __restrict__ mtxz) {
    int t = threadIdx.x, m = blockIdx.x;
    float z = gz[t];
    float z2 = z * z;
    int idx = m * 256 + t;
    mtxz[idx] = mtx[idx] * (z2 * z2);
}

// ---------------------------------------------------------------------------
// WMMA GEMM: C[256 x 16384] = A_f32[256 x 256] * B[256 x 16384]
// MODE 1: B = feture (f32), D = pad volume (writes float2(v,0))
// MODE 2: B = real parts of cropped pad volume, D = d_out
// block: 256 threads = 8 waves; block tile 16(M) x 128(N); K chunked by 32.
// ---------------------------------------------------------------------------
template<int MODE>
__global__ __launch_bounds__(256)
void lct_gemm_wmma(const float*  __restrict__ A,
                   const float*  __restrict__ Bf,
                   const float2* __restrict__ Bc,
                   float2*       __restrict__ Dpad,
                   float*        __restrict__ Dout) {
    __shared__ float As[16][34];    // row stride even -> 8B-aligned pairs
    __shared__ float Bs[32][130];

    const int tid  = threadIdx.x;
    const int lane = tid & 31;
    const int wv   = tid >> 5;          // wave 0..7 -> N sub-tile
    const int hlf  = lane >> 4;         // lane half (0/1)
    const int lm   = lane & 15;
    const int m0   = blockIdx.y * 16;
    const int n0b  = blockIdx.x * 128;

    v8f acc = {};

    for (int kb = 0; kb < 256; kb += 32) {
        // ---- stage A chunk 16x32 ----
        #pragma unroll
        for (int r = 0; r < 2; ++r) {
            int flat = tid + 256 * r;            // 0..511
            int mm = flat >> 5, kk = flat & 31;
            As[mm][kk] = A[(size_t)(m0 + mm) * 256 + kb + kk];
        }
        // ---- stage B chunk 32x128 (coalesced along n) ----
        #pragma unroll
        for (int r = 0; r < 16; ++r) {
            int flat = tid + 256 * r;            // 0..4095
            int k  = flat >> 7;                  // 0..31
            int nl = flat & 127;
            int nG = n0b + nl;
            float v;
            if constexpr (MODE == 1) {
                v = Bf[(size_t)(kb + k) * 16384 + nG];
            } else {
                int h = nG >> 7, w = nG & 127;
                v = Bc[(size_t)(kb + k) * 65536 + h * 256 + w].x;
            }
            Bs[k][nl] = v;
        }
        if (kb + 32 < 256) {
            if constexpr (MODE == 1)
                __builtin_prefetch(&Bf[(size_t)(kb + 32) * 16384 + n0b + (tid & 127)], 0, 0);
            else
                __builtin_prefetch(&Bc[(size_t)(kb + 32) * 65536 + ((n0b + (tid & 127)) >> 7) * 256], 0, 0);
        }
        __syncthreads();

#if LCT_HAVE_WMMA_F32
        // fp32 WMMA 16x16x4: A frag = {A[m][2h], A[m][2h+1]}, B frag likewise
        #pragma unroll
        for (int s8 = 0; s8 < 8; ++s8) {
            int ks = 4 * s8;
            v2f fa = *reinterpret_cast<const v2f*>(&As[lm][ks + 2 * hlf]);
            v2f fb;
            fb[0] = Bs[ks + 2 * hlf][wv * 16 + lm];
            fb[1] = Bs[ks + 2 * hlf + 1][wv * 16 + lm];
            acc = __builtin_amdgcn_wmma_f32_16x16x4_f32(
                      false, fa, false, fb, (short)0, acc, false, false);
        }
#else
        // fallback: f16 WMMA 16x16x32 (convert from f32 tiles at frag build)
        v16h fa, fb;
        #pragma unroll
        for (int q = 0; q < 8; ++q) {
            int ka = (q < 4) ? (8 * hlf + 2 * q) : (16 + 8 * hlf + 2 * (q - 4));
            fa[2 * q]     = (_Float16)As[lm][ka];
            fa[2 * q + 1] = (_Float16)As[lm][ka + 1];
            int kbb = (q < 4) ? (8 * hlf + 2 * q) : (16 + 8 * hlf + 2 * (q - 4));
            fb[2 * q]     = (_Float16)Bs[kbb][wv * 16 + lm];
            fb[2 * q + 1] = (_Float16)Bs[kbb + 1][wv * 16 + lm];
        }
        acc = __builtin_amdgcn_wmma_f32_16x16x32_f16(
                  false, fa, false, fb, (short)0, acc, false, false);
#endif
        __syncthreads();
    }

    // ---- epilogue: C layout VGPR r -> M = r + 8*hlf ----
    #pragma unroll
    for (int r = 0; r < 8; ++r) {
        int m = m0 + r + 8 * hlf;
        int p = n0b + wv * 16 + lm;
        if constexpr (MODE == 1) {
            int h = p >> 7, w = p & 127;
            Dpad[(size_t)m * 65536 + (size_t)h * 256 + w] = make_float2(acc[r], 0.f);
        } else {
            Dout[(size_t)m * 16384 + p] = acc[r];
        }
    }
}

// ---------------------------------------------------------------------------
// Radix-2 in-LDS FFT helpers
// ---------------------------------------------------------------------------
__device__ __forceinline__ float2 cmul_tw(float2 v, float c, float s) {
    return make_float2(v.x * c - v.y * s, v.x * s + v.y * c);
}

// FFT along x (contiguous, N=256); only (z<256, y<128) lines carry data.
// loadN: elements read from memory (rest zero); storeN: elements written.
__global__ __launch_bounds__(128)
void lct_fft_x(float2* __restrict__ d, float sign, int loadN, int storeN) {
    __shared__ float2 s[256];
    const int tid = threadIdx.x;
    const int li  = blockIdx.x;              // 0..32767
    const int z = li >> 7, y = li & 127;
    float2* base = d + (size_t)z * 65536 + (size_t)y * 256;
    #pragma unroll
    for (int q = 0; q < 2; ++q) {
        int i = tid + q * 128;
        s[i] = (i < loadN) ? base[i] : make_float2(0.f, 0.f);
    }
    __syncthreads();
    #pragma unroll
    for (int q = 0; q < 2; ++q) {
        int i = tid + q * 128;
        int j = (int)(__brev((unsigned)i) >> 24);
        if (j > i) { float2 t = s[i]; s[i] = s[j]; s[j] = t; }
    }
    __syncthreads();
    for (int len = 2; len <= 256; len <<= 1) {
        int hl = len >> 1;
        float ang = sign * 6.28318530717958647692f / (float)len;
        int b = tid;
        int off = b & (hl - 1);
        int i0 = ((b & ~(hl - 1)) << 1) | off;
        int i1 = i0 + hl;
        float sn, cs; __sincosf(ang * (float)off, &sn, &cs);
        float2 u = s[i0], w = cmul_tw(s[i1], cs, sn);
        s[i0] = make_float2(u.x + w.x, u.y + w.y);
        s[i1] = make_float2(u.x - w.x, u.y - w.y);
        __syncthreads();
    }
    #pragma unroll
    for (int q = 0; q < 2; ++q) {
        int i = tid + q * 128;
        if (i < storeN) base[i] = s[i];
    }
}

// FFT along y (stride 256, N=256), 16 lines (consecutive x) per block.
// Only z<256 planes are processed (blockIdx.y = z).
__global__ __launch_bounds__(256)
void lct_fft_y(float2* __restrict__ d, float sign, int loadN, int storeN) {
    __shared__ float2 s[16][257];
    const int tid  = threadIdx.x;
    const int line = tid & 15;
    const int sub  = tid >> 4;   // 0..15
    const size_t base = (size_t)blockIdx.y * 65536 + (size_t)blockIdx.x * 16 + line;
    #pragma unroll
    for (int k = 0; k < 16; ++k) {
        int e = sub * 16 + k;
        s[line][e] = (e < loadN) ? d[base + (size_t)e * 256] : make_float2(0.f, 0.f);
    }
    __syncthreads();
    #pragma unroll
    for (int k = 0; k < 16; ++k) {
        int i = sub * 16 + k;
        int j = (int)(__brev((unsigned)i) >> 24);
        if (j > i) { float2 t = s[line][i]; s[line][i] = s[line][j]; s[line][j] = t; }
    }
    __syncthreads();
    for (int len = 2; len <= 256; len <<= 1) {
        int hl = len >> 1;
        float ang = sign * 6.28318530717958647692f / (float)len;
        #pragma unroll
        for (int t = 0; t < 8; ++t) {
            int b = sub * 8 + t;                 // 128 butterflies / line
            int off = b & (hl - 1);
            int i0 = ((b & ~(hl - 1)) << 1) | off;
            int i1 = i0 + hl;
            float sn, cs; __sincosf(ang * (float)off, &sn, &cs);
            float2 u = s[line][i0], w = cmul_tw(s[line][i1], cs, sn);
            s[line][i0] = make_float2(u.x + w.x, u.y + w.y);
            s[line][i1] = make_float2(u.x - w.x, u.y - w.y);
        }
        __syncthreads();
    }
    #pragma unroll
    for (int k = 0; k < 16; ++k) {
        int e = sub * 16 + k;
        if (e < storeN) d[base + (size_t)e * 256] = s[line][e];
    }
}

// In-LDS 512-point FFT body for the fused z kernel (16 lines, 16 subs)
__device__ __forceinline__ void lct_fft512_lds(float2 (*s)[514], int line, int sub, float sign) {
    #pragma unroll
    for (int k = 0; k < 32; ++k) {
        int i = sub * 32 + k;
        int j = (int)(__brev((unsigned)i) >> 23);
        if (j > i) { float2 t = s[line][i]; s[line][i] = s[line][j]; s[line][j] = t; }
    }
    __syncthreads();
    for (int len = 2; len <= 512; len <<= 1) {
        int hl = len >> 1;
        float ang = sign * 6.28318530717958647692f / (float)len;
        #pragma unroll
        for (int t = 0; t < 16; ++t) {
            int b = sub * 16 + t;                // 256 butterflies / line
            int off = b & (hl - 1);
            int i0 = ((b & ~(hl - 1)) << 1) | off;
            int i1 = i0 + hl;
            float sn, cs; __sincosf(ang * (float)off, &sn, &cs);
            float2 u = s[line][i0], w = cmul_tw(s[line][i1], cs, sn);
            s[line][i0] = make_float2(u.x + w.x, u.y + w.y);
            s[line][i1] = make_float2(u.x - w.x, u.y - w.y);
        }
        __syncthreads();
    }
}

// Fused: fwd z-FFT (zero-extended to 512) -> Wiener+learned filter (-> *1/N)
//        -> inv z-FFT, storing only z<256 (all later consumers crop there).
// 16 lines (consecutive x) per block; blockIdx = (x-group, y).
__global__ __launch_bounds__(256)
void lct_fft_z_filter(float2* __restrict__ d, const float* __restrict__ lw,
                      const float* __restrict__ ipr, const float* __restrict__ ipi) {
    __shared__ float2 s[16][514];
    const int tid  = threadIdx.x;
    const int line = tid & 15;
    const int sub  = tid >> 4;   // 0..15
    const size_t base = (size_t)blockIdx.y * 256 + (size_t)blockIdx.x * 16 + line;

    #pragma unroll
    for (int k = 0; k < 32; ++k) {
        int e = sub * 32 + k;
        s[line][e] = (e < 256) ? d[base + (size_t)e * 65536] : make_float2(0.f, 0.f);
    }
    __syncthreads();

    lct_fft512_lds(s, line, sub, -1.f);          // forward

    const float invN = 1.0f / 33554432.0f;       // fold full ifftn 1/N here
    #pragma unroll
    for (int k = 0; k < 32; ++k) {
        int e = sub * 32 + k;
        size_t i = (size_t)e * 65536 + base;
        float wr = (lw[i]        + ipr[i]) * invN;
        float wi = (lw[i + NTOT] + ipi[i]) * invN;
        float2 f = s[line][e];
        s[line][e] = make_float2(f.x * wr - f.y * wi, f.x * wi + f.y * wr);
    }
    __syncthreads();

    lct_fft512_lds(s, line, sub, 1.f);           // inverse

    #pragma unroll
    for (int k = 0; k < 32; ++k) {
        int e = sub * 32 + k;
        if (e < 256) d[base + (size_t)e * 65536] = s[line][e];
    }
}

// ---------------------------------------------------------------------------
extern "C" void kernel_launch(void* const* d_in, const int* in_sizes, int n_in,
                              void* d_out, int out_size, void* d_ws, size_t ws_size,
                              hipStream_t stream) {
    const float* feture = (const float*)d_in[0];
    const float* gridz  = (const float*)d_in[1];
    const float* mtx    = (const float*)d_in[2];
    const float* mtxi   = (const float*)d_in[3];
    const float* ipr    = (const float*)d_in[4];
    const float* ipi    = (const float*)d_in[5];
    const float* lw     = (const float*)d_in[6];
    float* out = (float*)d_out;

    char* ws = (char*)d_ws;
    float2* pad  = (float2*)ws;                  // 268,435,456 B
    float*  mtxz = (float*)(ws + PADBYTES);      // 262,144 B

    lct_prep_mtx<<<256, 256, 0, stream>>>(mtx, gridz, mtxz);

    // scaled temporal-resampling GEMM -> (t<256, y<128, x<128) of pad volume
    lct_gemm_wmma<1><<<dim3(128, 16), 256, 0, stream>>>(mtxz, feture, nullptr, pad, nullptr);

    // forward FFT, exploiting zero-padding sparsity
    lct_fft_x<<<32768, 128, 0, stream>>>(pad, -1.f, 128, 256);          // z<256, y<128 lines
    lct_fft_y<<<dim3(16, 256), 256, 0, stream>>>(pad, -1.f, 128, 256);  // z<256 planes

    // fused fwd-z + filter + inv-z (stores only z<256)
    lct_fft_z_filter<<<dim3(16, 256), 256, 0, stream>>>(pad, lw, ipr, ipi);

    // inverse FFT, producing only the cropped region consumers need
    lct_fft_y<<<dim3(16, 256), 256, 0, stream>>>(pad, 1.f, 256, 128);   // store y<128
    lct_fft_x<<<32768, 128, 0, stream>>>(pad, 1.f, 256, 128);           // store x<128

    // inverse temporal-resampling GEMM on cropped real volume -> output
    lct_gemm_wmma<2><<<dim3(128, 16), 256, 0, stream>>>(mtxi, nullptr, pad, nullptr, out);
}
